// AttentionHead_3908420240000
// MI455X (gfx1250) — matrix operand order repaired and checked by reading, not compiled, not run on hardware
//
#include <hip/hip_runtime.h>

// ---------------------------------------------------------------------------
// Single-head causal attention (B=4, T=4096, D=1024, H=64), fp32 in/out.
// bf16 WMMA (v_wmma_f32_16x16x32_bf16), fp32 accumulation.
//   k1: x fp32 -> bf16
//   k2: Wq/Wk/Wv fp32 -> bf16, transposed to [3][H][D]
//   k3: fused QKV projection (one A-frag -> 3 WMMAs); q,k row-major bf16,
//       v transposed [B][H][T] bf16 (packed b128 stores)
//   k4: flash attention, block-uniform key loop; K/V tiles staged in LDS via
//       global_load_async_to_lds_b128 (ASYNCcnt) and shared by all 8 waves;
//       online softmax with wave32 shfl_xor; P transposed C->A through LDS
//       with s_wait_dscnt. Uniform loop costs ~3% extra WMMAs but cuts K/V
//       global traffic 8x.
// Workspace needed: ~40.3 MB.
// ---------------------------------------------------------------------------

typedef __attribute__((ext_vector_type(16))) __bf16          v16bf;
typedef __attribute__((ext_vector_type(8)))  float           v8f;
typedef __attribute__((ext_vector_type(8)))  unsigned short  ushort8;
typedef __attribute__((ext_vector_type(4)))  unsigned short  ushort4v;

constexpr int Bn = 4;
constexpr int Tn = 4096;
constexpr int Dn = 1024;
constexpr int Hn = 64;
constexpr float SCALE = 0.03125f;     // 1024^-0.5 (reference scales by embed dim)
constexpr float NEG_BIG = -3.0e38f;

union FragA { v16bf v; ushort8 u[2]; };   // 16x32 bf16 A fragment (8 VGPRs)
union FragB { v16bf v; ushort8 u[2]; };   // 32x16 bf16 B fragment (8 VGPRs)

__device__ __forceinline__ unsigned short f2bf(float f) {
    union { float f; unsigned u; } c; c.f = f;
    unsigned u = c.u;
    u += 0x7fffu + ((u >> 16) & 1u);      // round-to-nearest-even
    return (unsigned short)(u >> 16);
}

// async 16-byte global -> LDS copy (tracked on ASYNCcnt, no VGPR data)
__device__ __forceinline__ void async_ld16(unsigned lds_off, const void* g) {
    asm volatile("global_load_async_to_lds_b128 %0, %1, off"
                 :: "v"(lds_off), "v"(g) : "memory");
}

// ------------------------- k1: x fp32 -> bf16 ------------------------------
__global__ void ah_convert_x(const float* __restrict__ x,
                             unsigned short* __restrict__ xb) {
    size_t i = ((size_t)blockIdx.x * blockDim.x + threadIdx.x) * 4;
    float4 v = *(const float4*)(x + i);
    ushort4v o;
    o[0] = f2bf(v.x); o[1] = f2bf(v.y); o[2] = f2bf(v.z); o[3] = f2bf(v.w);
    *(ushort4v*)(xb + i) = o;
}

// --------------- k2: W[D][H] fp32 -> Wt[mat][H][D] bf16 --------------------
__global__ void ah_convert_w(const float* __restrict__ Wq,
                             const float* __restrict__ Wk,
                             const float* __restrict__ Wv,
                             unsigned short* __restrict__ wt) {
    int id = blockIdx.x * blockDim.x + threadIdx.x;
    if (id >= 3 * Dn * Hn) return;
    int mat = id / (Dn * Hn);
    int r   = id % (Dn * Hn);
    int h   = r / Dn;
    int d   = r % Dn;
    const float* W = (mat == 0) ? Wq : (mat == 1) ? Wk : Wv;
    wt[id] = f2bf(W[(size_t)d * Hn + h]);
}

// ------------------------- k3: fused QKV GEMM ------------------------------
__global__ void ah_qkv(const unsigned short* __restrict__ xb,
                       const unsigned short* __restrict__ wt,
                       const float* __restrict__ bq,
                       const float* __restrict__ bk,
                       const float* __restrict__ bv,
                       unsigned short* __restrict__ qo,
                       unsigned short* __restrict__ ko,
                       unsigned short* __restrict__ vto) {
    const int lane = threadIdx.x & 31;
    const int w    = threadIdx.x >> 5;
    const int rt   = w >> 2;            // 0..1
    const int ct   = w & 3;             // 0..3
    const int half = lane >> 4;         // 0..1
    const int nl   = lane & 15;         // 0..15

    const int row = blockIdx.x * 32 + rt * 16 + nl;   // token (A-matrix row)
    const int h   = ct * 16 + nl;                     // output column

    const unsigned short* arow = xb + (size_t)row * Dn;
    const unsigned short* bwq  = wt + 0 * (size_t)Hn * Dn + (size_t)h * Dn;
    const unsigned short* bwk  = wt + 1 * (size_t)Hn * Dn + (size_t)h * Dn;
    const unsigned short* bwv  = wt + 2 * (size_t)Hn * Dn + (size_t)h * Dn;

    v8f aq = {}, ak = {}, av = {};

    for (int kk = 0; kk < Dn / 32; ++kk) {
        const int kb = kk * 32;
        FragA a;
        a.u[0] = *(const ushort8*)(arow + kb + half * 16);
        a.u[1] = *(const ushort8*)(arow + kb + half * 16 + 8);
        const int off = kb + half * 16;
        FragB fq, fk, fv;
        fq.u[0] = *(const ushort8*)(bwq + off);
        fq.u[1] = *(const ushort8*)(bwq + off + 8);
        fk.u[0] = *(const ushort8*)(bwk + off);
        fk.u[1] = *(const ushort8*)(bwk + off + 8);
        fv.u[0] = *(const ushort8*)(bwv + off);
        fv.u[1] = *(const ushort8*)(bwv + off + 8);

        aq = __builtin_amdgcn_wmma_f32_16x16x32_bf16(false, a.v, false, fq.v,
                                                     (short)0, aq, false, false);
        ak = __builtin_amdgcn_wmma_f32_16x16x32_bf16(false, a.v, false, fk.v,
                                                     (short)0, ak, false, false);
        av = __builtin_amdgcn_wmma_f32_16x16x32_bf16(false, a.v, false, fv.v,
                                                     (short)0, av, false, false);
    }

    const float biq = bq[h], bik = bk[h], biv = bv[h];
    const int mbase = blockIdx.x * 32 + rt * 16 + 8 * half;

#pragma unroll
    for (int r = 0; r < 8; ++r) {
        size_t o = (size_t)(mbase + r) * Hn + h;
        qo[o] = f2bf(aq[r] + biq);
        ko[o] = f2bf(ak[r] + bik);
    }
    const int bidx = (blockIdx.x * 32) / Tn;
    const int tb   = mbase - bidx * Tn;
    ushort8 pk;
#pragma unroll
    for (int r = 0; r < 8; ++r) pk[r] = f2bf(av[r] + biv);
    *(ushort8*)(vto + ((size_t)bidx * Hn + h) * Tn + tb) = pk;
}

// ------------------------- k4: flash attention -----------------------------
// Grid (T/128, B); 8 waves/block; wave w owns 16 query rows q0 = x*128 + w*16.
// Block-uniform key loop; K/V tiles async-staged into LDS each iteration.
__global__ void ah_attn(const unsigned short* __restrict__ qb,
                        const unsigned short* __restrict__ kbm,
                        const unsigned short* __restrict__ vtm,
                        float* __restrict__ out) {
    const int b    = blockIdx.y;
    const int tid  = threadIdx.x;
    const int lane = tid & 31;
    const int w    = tid >> 5;
    const int half = lane >> 4;
    const int nl   = lane & 15;
    const int q0   = blockIdx.x * 128 + w * 16;

    __shared__ unsigned short kTile[32][64];   // [key][h]   4 KB
    __shared__ unsigned short vTile[64][32];   // [h][key]   4 KB
    __shared__ unsigned short pl[8][16][32];   // per-wave P 8 KB

    // Q A-fragments: frag s covers h = s*32 .. s*32+31
    FragA qf[2];
    const unsigned short* qrow = qb + ((size_t)b * Tn + q0 + nl) * Hn;
#pragma unroll
    for (int s = 0; s < 2; ++s) {
        qf[s].u[0] = *(const ushort8*)(qrow + s * 32 + half * 16);
        qf[s].u[1] = *(const ushort8*)(qrow + s * 32 + half * 16 + 8);
    }

    float m_run[8], l_run[8];
    v8f acc[4] = {{}, {}, {}, {}};
#pragma unroll
    for (int r = 0; r < 8; ++r) { m_run[r] = NEG_BIG; l_run[r] = 0.f; }

    const int kmaxblk = (blockIdx.x + 1) * 128;   // keys needed by this block

    for (int kb0 = 0; kb0 < kmaxblk; kb0 += 32) {
        // ---- async stage K tile (32x64) and V tile (64x32) into LDS ----
        {
            const int kkey = tid >> 3, kseg = tid & 7;      // 32 x 8 x 16B
            unsigned koff = (unsigned)(size_t)&kTile[kkey][kseg * 8];
            async_ld16(koff,
                       kbm + ((size_t)b * Tn + kb0 + kkey) * Hn + kseg * 8);
            const int vh = tid >> 2, vseg = tid & 3;        // 64 x 4 x 16B
            unsigned voff = (unsigned)(size_t)&vTile[vh][vseg * 8];
            async_ld16(voff,
                       vtm + ((size_t)b * Hn + vh) * Tn + kb0 + vseg * 8);
        }
        asm volatile("s_wait_asynccnt 0" ::: "memory");
        __syncthreads();                                    // tiles visible

        if (kb0 <= q0 + 15) {   // wave-uniform: tile intersects causal range
            // ---- S = scale * Q K^T (two 16-col C tiles from LDS) ----
            v8f s0 = {}, s1 = {};
#pragma unroll
            for (int hs = 0; hs < 2; ++hs) {
                const int hoff = hs * 32 + half * 16;
                FragB k0, k1;
                k0.u[0] = *(const ushort8*)&kTile[nl][hoff];
                k0.u[1] = *(const ushort8*)&kTile[nl][hoff + 8];
                k1.u[0] = *(const ushort8*)&kTile[16 + nl][hoff];
                k1.u[1] = *(const ushort8*)&kTile[16 + nl][hoff + 8];
                s0 = __builtin_amdgcn_wmma_f32_16x16x32_bf16(
                         false, qf[hs].v, false, k0.v, (short)0, s0, false, false);
                s1 = __builtin_amdgcn_wmma_f32_16x16x32_bf16(
                         false, qf[hs].v, false, k1.v, (short)0, s1, false, false);
            }

            // ---- scale + causal mask + online softmax ----
            const bool needMask = (kb0 + 31 > q0);
            float p0[8], p1[8], rmx[8];
#pragma unroll
            for (int r = 0; r < 8; ++r) {
                float a0 = s0[r] * SCALE;
                float a1 = s1[r] * SCALE;
                if (needMask) {
                    const int qg = q0 + r + 8 * half;
                    if (kb0 + nl      > qg) a0 = NEG_BIG;
                    if (kb0 + 16 + nl > qg) a1 = NEG_BIG;
                }
                p0[r] = a0; p1[r] = a1;
                rmx[r] = fmaxf(a0, a1);
            }
#pragma unroll
            for (int r = 0; r < 8; ++r) {
#pragma unroll
                for (int off = 1; off < 16; off <<= 1)
                    rmx[r] = fmaxf(rmx[r], __shfl_xor(rmx[r], off, 32));
            }
            float alpha[8], rs[8];
#pragma unroll
            for (int r = 0; r < 8; ++r) {
                const float mnew = fmaxf(m_run[r], rmx[r]);
                alpha[r] = __expf(m_run[r] - mnew);
                m_run[r] = mnew;
                p0[r] = __expf(p0[r] - mnew);
                p1[r] = __expf(p1[r] - mnew);
                rs[r] = p0[r] + p1[r];
            }
#pragma unroll
            for (int r = 0; r < 8; ++r) {
#pragma unroll
                for (int off = 1; off < 16; off <<= 1)
                    rs[r] += __shfl_xor(rs[r], off, 32);
                l_run[r] = l_run[r] * alpha[r] + rs[r];
            }
#pragma unroll
            for (int t = 0; t < 4; ++t)
#pragma unroll
                for (int r = 0; r < 8; ++r) acc[t][r] *= alpha[r];

            // ---- P (C layout) -> LDS row-major bf16 -> A fragment ----
#pragma unroll
            for (int r = 0; r < 8; ++r) {
                const int m = r + 8 * half;
                pl[w][m][nl]      = f2bf(p0[r]);
                pl[w][m][nl + 16] = f2bf(p1[r]);
            }
            asm volatile("s_wait_dscnt 0" ::: "memory");
            FragA pf;
            const unsigned short* pb = &pl[w][nl][half * 16];
            pf.u[0] = *(const ushort8*)(pb);
            pf.u[1] = *(const ushort8*)(pb + 8);

            // ---- O += P * V (V tile rows give contiguous key runs) ----
#pragma unroll
            for (int ht = 0; ht < 4; ++ht) {
                FragB vf;
                vf.u[0] = *(const ushort8*)&vTile[ht * 16 + nl][half * 16];
                vf.u[1] = *(const ushort8*)&vTile[ht * 16 + nl][half * 16 + 8];
                acc[ht] = __builtin_amdgcn_wmma_f32_16x16x32_bf16(
                              false, pf.v, false, vf.v, (short)0, acc[ht],
                              false, false);
            }
        }
        __syncthreads();   // all waves done reading before next tile fill
    }

    // ---- normalize and store fp32 output ----
    float inv[8];
#pragma unroll
    for (int r = 0; r < 8; ++r) inv[r] = 1.0f / l_run[r];
#pragma unroll
    for (int ht = 0; ht < 4; ++ht)
#pragma unroll
        for (int r = 0; r < 8; ++r) {
            const int q = q0 + r + 8 * half;
            out[((size_t)b * Tn + q) * Hn + ht * 16 + nl] = acc[ht][r] * inv[r];
        }
}

// ---------------------------------------------------------------------------
extern "C" void kernel_launch(void* const* d_in, const int* in_sizes, int n_in,
                              void* d_out, int out_size, void* d_ws, size_t ws_size,
                              hipStream_t stream) {
    (void)in_sizes; (void)n_in; (void)out_size; (void)ws_size;
    const float* x  = (const float*)d_in[0];
    const float* Wq = (const float*)d_in[1];
    const float* bq = (const float*)d_in[2];
    const float* Wk = (const float*)d_in[3];
    const float* bk = (const float*)d_in[4];
    const float* Wv = (const float*)d_in[5];
    const float* bv = (const float*)d_in[6];
    float* out = (float*)d_out;

    char* ws = (char*)d_ws;
    unsigned short* xb  = (unsigned short*)ws;                 // B*T*D bf16
    ws += (size_t)Bn * Tn * Dn * 2;
    unsigned short* wt  = (unsigned short*)ws;                 // 3*H*D bf16
    ws += (size_t)3 * Hn * Dn * 2;
    unsigned short* qo  = (unsigned short*)ws;                 // B*T*H bf16
    ws += (size_t)Bn * Tn * Hn * 2;
    unsigned short* ko  = (unsigned short*)ws;                 // B*T*H bf16
    ws += (size_t)Bn * Tn * Hn * 2;
    unsigned short* vto = (unsigned short*)ws;                 // B*H*T bf16

    const int nX = Bn * Tn * Dn;                // 16,777,216
    ah_convert_x<<<nX / (256 * 4), 256, 0, stream>>>(x, xb);
    ah_convert_w<<<(3 * Dn * Hn + 255) / 256, 256, 0, stream>>>(Wq, Wk, Wv, wt);
    ah_qkv<<<(Bn * Tn) / 32, 256, 0, stream>>>(xb, wt, bq, bk, bv, qo, ko, vto);
    ah_attn<<<dim3(Tn / 128, Bn), 256, 0, stream>>>(qo, ko, vto, out);
}
